// AttentionModule_88433376625114
// MI455X (gfx1250) — matrix operand order repaired and checked by reading, not compiled, not run on hardware
//
#include <hip/hip_runtime.h>

#define CDIM 256
#define C8   32
#define HWN  4096

typedef __attribute__((ext_vector_type(8)))  float  v8f;
typedef __bf16 bf16;
typedef __attribute__((ext_vector_type(16))) __bf16 v16bf;
typedef __attribute__((ext_vector_type(8)))  __bf16 v8bf;

#if __has_builtin(__builtin_amdgcn_exp2f)
#define EXP2F(x) __builtin_amdgcn_exp2f(x)
#else
#define EXP2F(x) exp2f(x)
#endif

// ---------------------------------------------------------------------------
// Kernel 1: fused 1x1-conv projection + q/k/v heads (bandwidth-bound, VALU).
//   qT[b][n][e]  bf16, pre-scaled by log2(e) so softmax runs in base-2
//   kT[b][m][e]  bf16 (rows feed the WMMA A fragment of the score GEMM)
//   vv[b][c][m]  bf16 (rows feed the WMMA A fragment of the PV GEMM)
// ---------------------------------------------------------------------------
__global__ __launch_bounds__(256) void qkv_proj_kernel(
    const float* __restrict__ x,
    const float* __restrict__ conv_w, const float* __restrict__ conv_b,
    const float* __restrict__ q_w, const float* __restrict__ q_b,
    const float* __restrict__ k_w, const float* __restrict__ k_b,
    const float* __restrict__ v_w, const float* __restrict__ v_b,
    bf16* __restrict__ qT, bf16* __restrict__ kT, bf16* __restrict__ vv)
{
  const int n = blockIdx.x * blockDim.x + threadIdx.x;   // 0..4095
  const int b = blockIdx.y;                              // 0..3
  const float* xb = x + ((size_t)b * CDIM) * HWN + n;
  const float LOG2E = 1.4426950408889634f;

  float proj[C8];
#pragma unroll
  for (int d = 0; d < C8; ++d) proj[d] = conv_b[d];
  for (int c = 0; c < CDIM; ++c) {
    float xc = xb[(size_t)c * HWN];
#pragma unroll
    for (int d = 0; d < C8; ++d)
      proj[d] = fmaf(conv_w[d * CDIM + c], xc, proj[d]);   // conv_w uniform -> s_loads
  }

  bf16* qrow = qT + ((size_t)b * HWN + n) * C8;
  bf16* krow = kT + ((size_t)b * HWN + n) * C8;
  for (int e = 0; e < C8; ++e) {
    float aq = q_b[e], ak = k_b[e];
#pragma unroll
    for (int d = 0; d < C8; ++d) {
      aq = fmaf(q_w[e * C8 + d], proj[d], aq);
      ak = fmaf(k_w[e * C8 + d], proj[d], ak);
    }
    qrow[e] = (bf16)(aq * LOG2E);     // base-2 softmax domain
    krow[e] = (bf16)ak;
  }

  bf16* vcol = vv + ((size_t)b * CDIM) * HWN + n;
  for (int c = 0; c < CDIM; ++c) {
    float av = v_b[c];
#pragma unroll
    for (int d = 0; d < C8; ++d) av = fmaf(v_w[c * C8 + d], proj[d], av);
    vcol[(size_t)c * HWN] = (bf16)av;     // coalesced across n
  }
}

// ---------------------------------------------------------------------------
// Kernel 2: flash attention, transposed-score formulation.
//   S^T tile:  A = K rows (M=m), B = Q cols (N=n)  ->  D keyed (m=vgpr, n=lane)
//   softmax stats are per-lane scalars (n = lane%16); one xor-16 shuffle each
//   PV:        A = V rows (M=c), B = P (K=m)       ->  acc keyed (c=vgpr, n=lane)
// ---------------------------------------------------------------------------
__global__ __launch_bounds__(128) void flash_attn_kernel(
    const bf16* __restrict__ qT, const bf16* __restrict__ kT, const bf16* __restrict__ vv,
    const float* __restrict__ x, const float* __restrict__ gamma, float* __restrict__ out)
{
  __shared__ float ostage[4][32 * 16];   // per-wave epilogue transpose

  const int lane = threadIdx.x & 31;
  const int wave = threadIdx.x >> 5;
  const int col  = lane & 15;            // n (and A-row index) within 16
  const int hi   = lane >> 4;            // lane half
  const int b    = blockIdx.y;
  const int ntile = blockIdx.x * 4 + wave;
  const int nbase = ntile * 16;

  // Q B-fragment: lane holds column n = nbase+col; lanes 0-15: e 0-7 & 16-23, lanes 16-31: e 8-15 & 24-31
  const bf16* qr = qT + ((size_t)b * HWN + (nbase + col)) * C8;
  v16bf qfrag = __builtin_shufflevector(*(const v8bf*)(qr + hi * 8), *(const v8bf*)(qr + 16 + hi * 8), 0,1,2,3,4,5,6,7,8,9,10,11,12,13,14,15);

  v8f acc[16];
#pragma unroll
  for (int t = 0; t < 16; ++t) acc[t] = v8f{};
  float m2 = -3.0e38f;   // running max (base-2 domain), keyed n = nbase+col
  float l  = 0.0f;       // running sum, same key

  const bf16* kbase = kT + (size_t)b * HWN * C8;
  const bf16* vbase = vv + (size_t)b * CDIM * HWN;

  for (int mb = 0; mb < HWN; mb += 32) {
    // K A-fragments: lane holds row m; lanes 0-15: e 0-7 & 16-23, lanes 16-31: e 8-15 & 24-31
    const bf16* kr0 = kbase + (size_t)(mb + col) * C8;
    const bf16* kr1 = kbase + (size_t)(mb + 16 + col) * C8;
    v16bf ka0 = __builtin_shufflevector(*(const v8bf*)(kr0 + hi * 8),
                                        *(const v8bf*)(kr0 + 16 + hi * 8),
                                        0,1,2,3,4,5,6,7,8,9,10,11,12,13,14,15);
    v16bf ka1 = __builtin_shufflevector(*(const v8bf*)(kr1 + hi * 8),
                                        *(const v8bf*)(kr1 + 16 + hi * 8),
                                        0,1,2,3,4,5,6,7,8,9,10,11,12,13,14,15);
    v8f z = v8f{};
    // sT0[r]: m = mb + r + 8*hi ; sT1[r]: m = mb + 16 + r + 8*hi ; n = nbase+col
    v8f s0 = __builtin_amdgcn_wmma_f32_16x16x32_bf16(false, ka0, false, qfrag, (short)0, z, false, false);
    v8f s1 = __builtin_amdgcn_wmma_f32_16x16x32_bf16(false, ka1, false, qfrag, (short)0, z, false, false);

    // chunk max over the 16 values this lane owns, then one cross-half exchange
    float c0 = fmaxf(fmaxf(fmaxf(s0[0], s0[1]), fmaxf(s0[2], s0[3])),
                     fmaxf(fmaxf(s0[4], s0[5]), fmaxf(s0[6], s0[7])));
    float c1 = fmaxf(fmaxf(fmaxf(s1[0], s1[1]), fmaxf(s1[2], s1[3])),
                     fmaxf(fmaxf(s1[4], s1[5]), fmaxf(s1[6], s1[7])));
    float cmax = fmaxf(c0, c1);
    cmax = fmaxf(cmax, __shfl_xor(cmax, 16, 32));

    float nm = fmaxf(m2, cmax);
    float sc = EXP2F(m2 - nm);
    m2 = nm;

    v8f p0, p1;
#pragma unroll
    for (int r = 0; r < 8; ++r) {
      p0[r] = EXP2F(s0[r] - nm);
      p1[r] = EXP2F(s1[r] - nm);
    }
    float t0 = (p0[0] + p0[1]) + (p0[2] + p0[3]);
    float t1 = (p0[4] + p0[5]) + (p0[6] + p0[7]);
    float t2 = (p1[0] + p1[1]) + (p1[2] + p1[3]);
    float t3 = (p1[4] + p1[5]) + (p1[6] + p1[7]);
    float csum = (t0 + t1) + (t2 + t3);
    csum += __shfl_xor(csum, 16, 32);
    l = l * sc + csum;

    // rescale accumulators only when the running max actually moved
    if (__any(sc < 1.0f)) {
#pragma unroll
      for (int t = 0; t < 16; ++t)
#pragma unroll
        for (int r = 0; r < 8; ++r) acc[t][r] *= sc;
    }

    // Build P B-fragment: lane-half h needs K=m elements {16h..16h+15} for column n.
    // Own half supplies 8 of them; the partner half (lane^16) supplies the other 8.
    v8bf p0b, p1b;
#pragma unroll
    for (int r = 0; r < 8; ++r) { p0b[r] = (bf16)p0[r]; p1b[r] = (bf16)p1[r]; }
    int4 own0 = __builtin_bit_cast(int4, p0b);
    int4 own1 = __builtin_bit_cast(int4, p1b);
    int4 send = hi ? own0 : own1;          // give partner what it lacks
    int4 recv;
    recv.x = __shfl_xor(send.x, 16, 32);
    recv.y = __shfl_xor(send.y, 16, 32);
    recv.z = __shfl_xor(send.z, 16, 32);
    recv.w = __shfl_xor(send.w, 16, 32);
    int4 lo4 = hi ? recv : own0;           // elems 0-7 : m = 16h .. 16h+7
    int4 hi4 = hi ? own1 : recv;           // elems 8-15: m = 16h+8 .. 16h+15
    v16bf pfrag = __builtin_shufflevector(__builtin_bit_cast(v8bf, lo4),
                                          __builtin_bit_cast(v8bf, hi4),
                                          0,1,2,3,4,5,6,7,8,9,10,11,12,13,14,15);

    // PV: A = V rows (contiguous 16B loads), B = P
#pragma unroll
    for (int t = 0; t < 16; ++t) {
      const bf16* vr = vbase + (size_t)(t * 16 + col) * HWN + mb;
      v16bf vfrag = __builtin_shufflevector(*(const v8bf*)(vr + hi * 16),
                                            *(const v8bf*)(vr + hi * 16 + 8),
                                            0,1,2,3,4,5,6,7,8,9,10,11,12,13,14,15);
      acc[t] = __builtin_amdgcn_wmma_f32_16x16x32_bf16(false, vfrag, false, pfrag, (short)0, acc[t], false, false);
    }
  }

  const float inv = 1.0f / l;          // keyed n = nbase+col
  const float g = gamma[0];
  float* os = ostage[wave];
#pragma unroll
  for (int tp = 0; tp < 8; ++tp) {
    asm volatile("s_wait_dscnt 0" ::: "memory");   // prior pair's LDS reads done
#pragma unroll
    for (int r = 0; r < 8; ++r) {
      // acc[t][r] is out_nT[c = 16t + r + 8*hi][n = nbase+col]
      os[(r + 8 * hi) * 16 + col]      = acc[2 * tp][r]     * inv;
      os[(16 + r + 8 * hi) * 16 + col] = acc[2 * tp + 1][r] * inv;
    }
    asm volatile("s_wait_dscnt 0" ::: "memory");   // stores visible before reads
    const int c = tp * 32 + lane;                  // each lane owns one channel row
    const size_t off = ((size_t)b * CDIM + c) * HWN + nbase;
    const float4* xp = (const float4*)(x + off);
    float4* op = (float4*)(out + off);
#pragma unroll
    for (int j = 0; j < 4; ++j) {
      float4 xv = xp[j];
      float4 ov;
      ov.x = g * os[lane * 16 + 4 * j + 0] + xv.x;
      ov.y = g * os[lane * 16 + 4 * j + 1] + xv.y;
      ov.z = g * os[lane * 16 + 4 * j + 2] + xv.z;
      ov.w = g * os[lane * 16 + 4 * j + 3] + xv.w;
      op[j] = ov;
    }
  }
}

extern "C" void kernel_launch(void* const* d_in, const int* in_sizes, int n_in,
                              void* d_out, int out_size, void* d_ws, size_t ws_size,
                              hipStream_t stream) {
  (void)in_sizes; (void)n_in; (void)out_size; (void)ws_size;
  const float* x      = (const float*)d_in[0];
  const float* conv_w = (const float*)d_in[1];
  const float* conv_b = (const float*)d_in[2];
  const float* q_w    = (const float*)d_in[3];
  const float* q_b    = (const float*)d_in[4];
  const float* k_w    = (const float*)d_in[5];
  const float* k_b    = (const float*)d_in[6];
  const float* v_w    = (const float*)d_in[7];
  const float* v_b    = (const float*)d_in[8];
  const float* gamma  = (const float*)d_in[9];
  float* out = (float*)d_out;

  // Workspace: qT (1 MB) | kT (1 MB) | vv (8 MB), all bf16
  bf16* qT = (bf16*)d_ws;
  bf16* kT = qT + (size_t)4 * HWN * C8;
  bf16* vv = kT + (size_t)4 * HWN * C8;

  qkv_proj_kernel<<<dim3(HWN / 256, 4), 256, 0, stream>>>(
      x, conv_w, conv_b, q_w, q_b, k_w, k_b, v_w, v_b, qT, kT, vv);

  flash_attn_kernel<<<dim3(HWN / 16 / 4, 4), 128, 0, stream>>>(
      qT, kT, vv, x, gamma, out);
}
